// HungarianMatcher_67181878444274
// MI455X (gfx1250) — matrix operand order repaired and checked by reading, not compiled, not run on hardware
//
#include <hip/hip_runtime.h>
#include <hip/hip_bf16.h>

typedef __attribute__((ext_vector_type(16))) _Float16 v16h;
typedef __attribute__((ext_vector_type(8)))  float    v8f;
typedef __attribute__((ext_vector_type(4)))  float    v4f;

#define NQ 400
#define NP 300000
#define NG 50
#define NC 19
#define GPAD 64
#define NKB (NP / 32)                              /* 9375 k-blocks, NP % 32 == 0 */
#define KWAVE 1024
#define NWAVES 8
#define BLOCK_PTS (KWAVE * NWAVES)                 /* 8192 */
#define PCHUNKS ((NP + BLOCK_PTS - 1) / BLOCK_PTS) /* 37 */
#define QTILES (NQ / 16)                           /* 25 */

// sigmoid via gfx1250 v_tanh_f32 (1 trans + 2 valu) with raw-rcp fallback
__device__ __forceinline__ float fast_sigmoid(float x) {
#if defined(__HIP_DEVICE_COMPILE__) && __has_builtin(__builtin_amdgcn_tanhf)
    return __builtin_amdgcn_tanhf(0.5f * x) * 0.5f + 0.5f;
#elif defined(__HIP_DEVICE_COMPILE__)
    return __builtin_amdgcn_rcpf(1.0f + __expf(-x));
#else
    (void)x; return 0.0f;
#endif
}

// ---------------------------------------------------------------- zero ws
__global__ void hm_zero_kernel(float* ws, int n) {
    int i = blockIdx.x * blockDim.x + threadIdx.x;
    if (i < n) ws[i] = 0.0f;
}

// ---------------------------------------------------------------- label histogram -> counts[g] (float)
__global__ void hm_hist_kernel(const int* __restrict__ labels, float* __restrict__ counts) {
    __shared__ float bins[GPAD];
    if (threadIdx.x < GPAD) bins[threadIdx.x] = 0.0f;
    __syncthreads();
    for (int i = blockIdx.x * blockDim.x + threadIdx.x; i < NP; i += gridDim.x * blockDim.x) {
        atomicAdd(&bins[labels[i]], 1.0f);          // ds_add_f32
    }
    __syncthreads();
    if (threadIdx.x < GPAD) {
        float v = bins[threadIdx.x];
        if (v != 0.0f) atomicAdd(&counts[threadIdx.x], v);
    }
}

// ---------------------------------------------------------------- precompute one-hot B fragments
// wsB layout: [kb][tile 0..3][lane 0..31] x 16 halves (32B per lane, lane-contiguous -> 1KB coalesced)
__global__ __launch_bounds__(256, 1)
void hm_bfrag_kernel(const int* __restrict__ labels, _Float16* __restrict__ wsB) {
    const int lane = threadIdx.x & 31;
    const int wave = threadIdx.x >> 5;
    const int kb   = blockIdx.x * NWAVES + wave;
    if (kb >= NKB) return;

    const int k0     = kb * 32;
    const int kgroup = lane >> 4;
    const int col    = lane & 15;

    const int4* pL = (const int4*)(labels + k0 + kgroup * 16);
    int4 l0 = pL[0], l1 = pL[1], l2 = pL[2], l3 = pL[3];
    int lab[16] = { l0.x, l0.y, l0.z, l0.w,  l1.x, l1.y, l1.z, l1.w,
                    l2.x, l2.y, l2.z, l2.w,  l3.x, l3.y, l3.z, l3.w };

#pragma unroll
    for (int t = 0; t < 4; ++t) {
        const int g = t * 16 + col;
        v16h b;
#pragma unroll
        for (int j = 0; j < 16; ++j) b[j] = (lab[j] == g) ? (_Float16)1.0f : (_Float16)0.0f;
        *(v16h*)(wsB + ((size_t)(kb * 4 + t) * 32 + lane) * 16) = b;
    }
}

// ---------------------------------------------------------------- pipeline stage helpers (fast path)
struct HmBufs {
    v4f  x0, x1, x2, x3;        // 16 logits floats (A operand, pre-sigmoid)
    v16h b0, b1, b2, b3;        // 4 one-hot B fragments
};

__device__ __forceinline__ void hm_load_step(HmBufs& B, const float* rowPtr,
                                             const _Float16* __restrict__ wsB,
                                             unsigned k0, int halfSel, int lane) {
    // A: non-temporal (streamed once; keep L2 for B fragments)
    const v4f* pA0 = (const v4f*)(rowPtr + k0 + halfSel * 16);
    const v4f* pA1 = (const v4f*)(rowPtr + k0 + halfSel * 16 + 8);
    B.x0 = __builtin_nontemporal_load(pA0);
    B.x1 = __builtin_nontemporal_load(pA0 + 1);
    B.x2 = __builtin_nontemporal_load(pA1);
    B.x3 = __builtin_nontemporal_load(pA1 + 1);
    // B fragments: regular temporal loads (L2-resident after first q-tile)
    const v16h* pB = (const v16h*)(wsB + ((size_t)(k0 >> 5) * 128 + (unsigned)lane) * 16);
    B.b0 = pB[0 * 32];
    B.b1 = pB[1 * 32];
    B.b2 = pB[2 * 32];
    B.b3 = pB[3 * 32];
}

__device__ __forceinline__ void hm_compute_step(const HmBufs& B,
                                                v8f& c0, v8f& c1, v8f& c2, v8f& c3) {
    float xs[16] = { B.x0.x, B.x0.y, B.x0.z, B.x0.w,  B.x1.x, B.x1.y, B.x1.z, B.x1.w,
                     B.x2.x, B.x2.y, B.x2.z, B.x2.w,  B.x3.x, B.x3.y, B.x3.z, B.x3.w };
    v16h a;
#pragma unroll
    for (int j = 0; j < 16; ++j) a[j] = (_Float16)fast_sigmoid(xs[j]);

    c0 = __builtin_amdgcn_wmma_f32_16x16x32_f16(false, a, false, B.b0, (short)0, c0, false, false);
    c1 = __builtin_amdgcn_wmma_f32_16x16x32_f16(false, a, false, B.b1, (short)0, c1, false, false);
    c2 = __builtin_amdgcn_wmma_f32_16x16x32_f16(false, a, false, B.b2, (short)0, c2, false, false);
    c3 = __builtin_amdgcn_wmma_f32_16x16x32_f16(false, a, false, B.b3, (short)0, c3, false, false);
}

// ---------------------------------------------------------------- main (fast path): double-buffered pipeline
__global__ __launch_bounds__(256, 1)
void hm_dice_wmma_pre(const float* __restrict__ logits,
                      const _Float16* __restrict__ wsB,
                      float* __restrict__ S) {
    __shared__ float tile[16 * GPAD];

    const int tid  = threadIdx.x;
    const int lane = tid & 31;
    const int wave = tid >> 5;

    for (int i = tid; i < 16 * GPAD; i += 256) tile[i] = 0.0f;
    __syncthreads();

    const int qbase   = blockIdx.x * 16;
    const int row     = qbase + (lane & 15);
    const int halfSel = lane >> 4;
    const int col     = lane & 15;
    const float* rowPtr = logits + (size_t)row * (size_t)NP;

    // wave-uniform scalars: force into SGPRs so all loop control is scalar branches
    const int pbase = __builtin_amdgcn_readfirstlane(blockIdx.y * BLOCK_PTS + wave * KWAVE);
    int kend = NP - pbase;
    if (kend > KWAVE) kend = KWAVE;
    const int steps = (kend > 0) ? (kend >> 5) : 0;

    v8f c0 = {}, c1 = {}, c2 = {}, c3 = {};

    if (steps > 0) {
        HmBufs A, Bn;
        hm_load_step(A, rowPtr, wsB, (unsigned)pbase, halfSel, lane);
        int i = 0;
        for (;;) {
            // branchless prefetch: clamp past-the-end prefetch back to pbase (valid, discarded)
            unsigned kn = (unsigned)((i + 1 < steps) ? (pbase + (i + 1) * 32) : pbase);
            hm_load_step(Bn, rowPtr, wsB, kn, halfSel, lane);
            hm_compute_step(A, c0, c1, c2, c3);
            ++i;
            if (i >= steps) break;
            kn = (unsigned)((i + 1 < steps) ? (pbase + (i + 1) * 32) : pbase);
            hm_load_step(A, rowPtr, wsB, kn, halfSel, lane);
            hm_compute_step(Bn, c0, c1, c2, c3);
            ++i;
            if (i >= steps) break;
        }
    }

    const int mhi = halfSel * 8;
#pragma unroll
    for (int r = 0; r < 8; ++r) {
        const int m = mhi + r;
        atomicAdd(&tile[m * GPAD + col +  0], c0[r]);
        atomicAdd(&tile[m * GPAD + col + 16], c1[r]);
        atomicAdd(&tile[m * GPAD + col + 32], c2[r]);
        atomicAdd(&tile[m * GPAD + col + 48], c3[r]);
    }
    __syncthreads();

    for (int i = tid; i < 16 * GPAD; i += 256) {
        const int m = i >> 6;
        const int g = i & (GPAD - 1);
        atomicAdd(&S[(size_t)(qbase + m) * GPAD + g], tile[i]);
    }
}

// ---------------------------------------------------------------- main (fallback): build B on the fly
__global__ __launch_bounds__(256, 1)
void hm_dice_wmma_fly(const float* __restrict__ logits,
                      const int*   __restrict__ labels,
                      float*       __restrict__ S) {
    __shared__ float tile[16 * GPAD];

    const int tid  = threadIdx.x;
    const int lane = tid & 31;
    const int wave = tid >> 5;

    for (int i = tid; i < 16 * GPAD; i += 256) tile[i] = 0.0f;
    __syncthreads();

    const int qbase   = blockIdx.x * 16;
    const int row     = qbase + (lane & 15);
    const int halfSel = lane >> 4;
    const size_t rowOff = (size_t)row * (size_t)NP;
    const int pbase = __builtin_amdgcn_readfirstlane(blockIdx.y * BLOCK_PTS + wave * KWAVE);
    const int col   = lane & 15;

    int kend = NP - pbase;
    if (kend > KWAVE) kend = KWAVE;

    v8f c0 = {}, c1 = {}, c2 = {}, c3 = {};

    for (int kk = 0; kk < kend; kk += 32) {
        const int k0 = pbase + kk;

        const v4f* pA0 = (const v4f*)(logits + rowOff + (size_t)(k0 + halfSel * 16));
        const v4f* pA1 = (const v4f*)(logits + rowOff + (size_t)(k0 + halfSel * 16 + 8));
        v4f x0 = __builtin_nontemporal_load(pA0);
        v4f x1 = __builtin_nontemporal_load(pA0 + 1);
        v4f x2 = __builtin_nontemporal_load(pA1);
        v4f x3 = __builtin_nontemporal_load(pA1 + 1);
        float xs[16] = { x0.x, x0.y, x0.z, x0.w,  x1.x, x1.y, x1.z, x1.w,
                         x2.x, x2.y, x2.z, x2.w,  x3.x, x3.y, x3.z, x3.w };
        v16h a;
#pragma unroll
        for (int j = 0; j < 16; ++j) a[j] = (_Float16)fast_sigmoid(xs[j]);

        const int4* pL = (const int4*)(labels + k0 + halfSel * 16);
        int4 l0 = pL[0], l1 = pL[1], l2 = pL[2], l3 = pL[3];
        int lab[16] = { l0.x, l0.y, l0.z, l0.w,  l1.x, l1.y, l1.z, l1.w,
                        l2.x, l2.y, l2.z, l2.w,  l3.x, l3.y, l3.z, l3.w };

        v16h b;
#pragma unroll
        for (int j = 0; j < 16; ++j) b[j] = (lab[j] == col)      ? (_Float16)1.0f : (_Float16)0.0f;
        c0 = __builtin_amdgcn_wmma_f32_16x16x32_f16(false, a, false, b, (short)0, c0, false, false);
#pragma unroll
        for (int j = 0; j < 16; ++j) b[j] = (lab[j] == col + 16) ? (_Float16)1.0f : (_Float16)0.0f;
        c1 = __builtin_amdgcn_wmma_f32_16x16x32_f16(false, a, false, b, (short)0, c1, false, false);
#pragma unroll
        for (int j = 0; j < 16; ++j) b[j] = (lab[j] == col + 32) ? (_Float16)1.0f : (_Float16)0.0f;
        c2 = __builtin_amdgcn_wmma_f32_16x16x32_f16(false, a, false, b, (short)0, c2, false, false);
#pragma unroll
        for (int j = 0; j < 16; ++j) b[j] = (lab[j] == col + 48) ? (_Float16)1.0f : (_Float16)0.0f;
        c3 = __builtin_amdgcn_wmma_f32_16x16x32_f16(false, a, false, b, (short)0, c3, false, false);
    }

    const int mhi = halfSel * 8;
#pragma unroll
    for (int r = 0; r < 8; ++r) {
        const int m = mhi + r;
        atomicAdd(&tile[m * GPAD + col +  0], c0[r]);
        atomicAdd(&tile[m * GPAD + col + 16], c1[r]);
        atomicAdd(&tile[m * GPAD + col + 32], c2[r]);
        atomicAdd(&tile[m * GPAD + col + 48], c3[r]);
    }
    __syncthreads();

    for (int i = tid; i < 16 * GPAD; i += 256) {
        const int m = i >> 6;
        const int g = i & (GPAD - 1);
        atomicAdd(&S[(size_t)(qbase + m) * GPAD + g], tile[i]);
    }
}

// ---------------------------------------------------------------- epilogue
__global__ void hm_final_kernel(const float* __restrict__ cls_preds,
                                const float* __restrict__ conf,
                                const int*   __restrict__ cls_labels,
                                const float* __restrict__ S,
                                const float* __restrict__ counts,
                                float*       __restrict__ out) {
    const int q = blockIdx.x * blockDim.x + threadIdx.x;
    if (q >= NQ) return;

    float lg[NC];
    float mx = -3.402823466e38f;
#pragma unroll
    for (int c = 0; c < NC; ++c) { lg[c] = cls_preds[q * NC + c]; mx = fmaxf(mx, lg[c]); }
    float sum = 0.0f;
#pragma unroll
    for (int c = 0; c < NC; ++c) { lg[c] = __expf(lg[c] - mx); sum += lg[c]; }
    const float inv = 1.0f / sum;

    float rowsum = 0.0f;
    float Srow[NG];
#pragma unroll 10
    for (int g = 0; g < NG; ++g) { Srow[g] = S[(size_t)q * GPAD + g]; rowsum += Srow[g]; }

    const float cf = conf[q];
#pragma unroll 10
    for (int g = 0; g < NG; ++g) {
        const float num  = 2.0f * Srow[g];
        const float den  = rowsum + counts[g];
        const float dice = 1.0f - (num + 1.0f) / (den + 1.0f);
        const float clsc = -lg[cls_labels[g]] * inv;
        out[q * NG + g] = clsc + dice - cf;
    }
}

// ----------------------------------------------------------------
extern "C" void kernel_launch(void* const* d_in, const int* in_sizes, int n_in,
                              void* d_out, int out_size, void* d_ws, size_t ws_size,
                              hipStream_t stream) {
    const float* cls_preds   = (const float*)d_in[0];  // [1,400,19]
    const float* mask_logits = (const float*)d_in[1];  // [1,400,300000]
    const float* conf_preds  = (const float*)d_in[2];  // [1,400]
    const int*   inst_labels = (const int*)  d_in[3];  // [300000]
    const int*   cls_labels  = (const int*)  d_in[4];  // [50]
    float* out = (float*)d_out;                        // [400,50]

    float* wsS   = (float*)d_ws;                        // [400*64]
    float* wsCnt = wsS + NQ * GPAD;                     // [64]
    const int nzero = NQ * GPAD + GPAD;

    size_t headBytes = ((size_t)nzero * sizeof(float) + 255) & ~(size_t)255;
    const size_t bBytes = (size_t)NKB * 4 * 32 * 16 * sizeof(_Float16);   // 38.4 MB
    _Float16* wsB = (_Float16*)((char*)d_ws + headBytes);
    const bool fast = (ws_size >= headBytes + bBytes);

    hm_zero_kernel<<<(nzero + 255) / 256, 256, 0, stream>>>(wsS, nzero);
    hm_hist_kernel<<<512, 256, 0, stream>>>(inst_labels, wsCnt);

    dim3 grid(QTILES, PCHUNKS);
    if (fast) {
        hm_bfrag_kernel<<<(NKB + NWAVES - 1) / NWAVES, 256, 0, stream>>>(inst_labels, wsB);
        hm_dice_wmma_pre<<<grid, 256, 0, stream>>>(mask_logits, wsB, wsS);
    } else {
        hm_dice_wmma_fly<<<grid, 256, 0, stream>>>(mask_logits, inst_labels, wsS);
    }

    hm_final_kernel<<<(NQ + 255) / 256, 256, 0, stream>>>(cls_preds, conf_preds, cls_labels,
                                                          wsS, wsCnt, out);
}
